// Gnn_basic_867583394363
// MI455X (gfx1250) — matrix operand order repaired and checked by reading, not compiled, not run on hardware
//
#include <hip/hip_runtime.h>

// ---------------------------------------------------------------------------
// GNN (GraphNet-style) forward for MI455X / gfx1250.
// v_wmma_f32_16x16x32_bf16 for all GEMMs. Weights pre-packed per block into
// LDS in per-lane B-fragment order (bf16); activations bounced through LDS in
// per-lane A-fragment order (bf16). h lives in the workspace as bf16 and
// stays L2-resident, so edge gathers and segment-sum atomics avoid HBM.
// ---------------------------------------------------------------------------

typedef __attribute__((ext_vector_type(16))) __bf16 v16bf;
typedef __attribute__((ext_vector_type(8)))  __bf16 v8bf;
typedef __attribute__((ext_vector_type(8)))  float  v8f;

#define DEV static __device__ __forceinline__

DEV v8f wmma_bf16(v16bf a, v16bf b, v8f c) {
  return __builtin_amdgcn_wmma_f32_16x16x32_bf16(false, a, false, b,
                                                 (short)0, c, false, false);
}

DEV void lds_fence()   { asm volatile("s_wait_dscnt 0x0" ::: "memory"); }
DEV void sched_fence() { asm volatile("" ::: "memory"); }

// ---- A fragment from f32 global data (K=16 input, zero-padded to 32) ------
DEV v16bf load_a_pad16(const float* x, int ld, int lane) {
  int m  = lane & 15;
  int kb = (lane >> 4) << 3;
  const float* r = x + m * ld;
  v16bf a;
#pragma unroll
  for (int j = 0; j < 8; ++j) a[j] = (__bf16)r[kb + j];
#pragma unroll
  for (int i = 8; i < 16; ++i) a[i] = (__bf16)0.0f;
  return a;
}

// ---- A fragment from f32 row (caller passes row-m pointer + k0) -----------
DEV v16bf load_a_row(const float* r, int lane) {
  int kb = (lane >> 4) << 3;
  v16bf a;
#pragma unroll
  for (int j = 0; j < 8; ++j) a[j]     = (__bf16)r[kb + j];
#pragma unroll
  for (int j = 0; j < 8; ++j) a[8 + j] = (__bf16)r[16 + kb + j];
  return a;
}

// ---- A fragment from bf16 row-major row: two contiguous 16B chunks --------
DEV v16bf load_a_bf16row(const __bf16* r, int lane) {
  int kb = (lane >> 4) << 3;
  v8bf lo = *(const v8bf*)(r + kb);
  v8bf hi = *(const v8bf*)(r + 16 + kb);
  return __builtin_shufflevector(lo, hi, 0, 1, 2, 3, 4, 5, 6, 7,
                                 8, 9, 10, 11, 12, 13, 14, 15);
}

// ---- Packed B fragments ---------------------------------------------------
// Fragment f = kb*ntiles + nt; lane's 16 bf16 stored contiguously (32B).
DEV v16bf load_bpk(const __bf16* pk, int frag, int lane) {
  return *(const v16bf*)(pk + (((frag << 5) + lane) << 4));
}

// Pack W[Kreal x (ntiles*16)] (row-major f32, global) into B-fragment order.
// Rows Kreal..Kpad-1 are zeros (Kpad multiple of 32).
DEV void pack_b(__bf16* dst, const float* __restrict__ W, int Kreal, int Kpad,
                int ntiles, int tid, int nth) {
  int ldn = ntiles << 4;
  int total = Kpad * ldn;
  for (int e = tid; e < total; e += nth) {
    int j    = e & 15;
    int lane = (e >> 4) & 31;
    int frag = e >> 9;
    int kb   = frag / ntiles;
    int nt   = frag - kb * ntiles;
    int k    = (kb << 5) + ((j >> 3) << 4) + ((lane >> 4) << 3) + (j & 7);
    int n    = (nt << 4) + (lane & 15);
    dst[e] = (__bf16)((k < Kreal) ? W[k * ldn + n] : 0.0f);
  }
}

// ---- Packed A activations (16x64 tile -> 2 K-blocks) ----------------------
DEV v16bf load_apk(const __bf16* ap, int kb, int lane) {
  return *(const v16bf*)(ap + (((kb << 5) + lane) << 4));
}

// Store D-tile element (row m, col c of the 16x64 tile) into A-pack order.
DEV void apack_store(__bf16* ap, int m, int c, float v) {
  int kb     = c >> 5;
  int ko     = c & 31;
  int lane_t = m + (((ko >> 3) & 1) << 4);
  int j      = ((ko >> 4) << 3) + (ko & 7);
  ap[(((kb << 5) + lane_t) << 4) + j] = (__bf16)v;
}

DEV void stage(float* dst, const float* __restrict__ src, int n, int tid, int nth) {
  for (int i = tid; i < n; i += nth) dst[i] = src[i];
}

// ===========================================================================
__global__ void gnn_zero_kernel(float* __restrict__ p, long n) {
  long i = (long)blockIdx.x * blockDim.x + threadIdx.x;
  if (i < n) p[i] = 0.0f;
}

// ===========================================================================
// Kernel 1: node encoder  [N,16] -> relu(X W0 + b0) W1 + b1 -> h (bf16 [N,64])
// ===========================================================================
__global__ __launch_bounds__(256) void gnn_enc_kernel(
    const float* __restrict__ X, const float* __restrict__ W0,
    const float* __restrict__ B0, const float* __restrict__ W1,
    const float* __restrict__ B1, __bf16* __restrict__ Y, int rows) {
  __shared__ alignas(32) __bf16 bW0[4 * 512];   // K=32(pad), 4 ntiles
  __shared__ alignas(32) __bf16 bW1[8 * 512];   // K=64, 4 ntiles
  __shared__ alignas(32) __bf16 aPk[8][1024];   // per-wave A-pack
  __shared__ float sB0[64], sB1[64];
  int tid = threadIdx.x, nth = blockDim.x;
  const float* Xb = X + (size_t)blockIdx.y * rows * 16;
  __bf16*      Yb = Y + (size_t)blockIdx.y * rows * 64;
  pack_b(bW0, W0, 16, 32, 4, tid, nth);
  pack_b(bW1, W1, 64, 64, 4, tid, nth);
  stage(sB0, B0, 64, tid, nth);
  stage(sB1, B1, 64, tid, nth);
  __syncthreads();

  int wave = tid >> 5, lane = tid & 31;
  int tile = blockIdx.x * 8 + wave;
  if (tile * 16 >= rows) return;        // wave-uniform exit: EXEC all-1s
  int n = lane & 15, hi = lane >> 4;
  __bf16* ap = aPk[wave];

  v16bf a0 = load_a_pad16(Xb + (size_t)tile * 256, 16, lane);
#pragma unroll
  for (int nt = 0; nt < 4; ++nt) {
    v8f acc = {};
    acc = wmma_bf16(a0, load_bpk(bW0, nt, lane), acc);
    float bias = sB0[nt * 16 + n];
#pragma unroll
    for (int r = 0; r < 8; ++r) {
      float v = acc[r] + bias;
      apack_store(ap, r + 8 * hi, nt * 16 + n, v > 0.0f ? v : 0.0f);
    }
  }
  lds_fence();

  v16bf h0 = load_apk(ap, 0, lane);
  v16bf h1 = load_apk(ap, 1, lane);
  sched_fence();
  __bf16* yt = Yb + (size_t)tile * 16 * 64;
#pragma unroll
  for (int nt = 0; nt < 4; ++nt) {
    v8f acc = {};
    acc = wmma_bf16(h0, load_bpk(bW1, nt, lane), acc);
    acc = wmma_bf16(h1, load_bpk(bW1, 4 + nt, lane), acc);
    float bias = sB1[nt * 16 + n];
#pragma unroll
    for (int r = 0; r < 8; ++r)
      yt[(r + 8 * hi) * 64 + nt * 16 + n] = (__bf16)(acc[r] + bias);
  }
}

// ===========================================================================
// Kernel 2 (fused): edge_enc -> gather h[e0],h[e1] -> edge_proc ->
// atomic scatter-add into agg[e0].  ee never touches HBM.
// ===========================================================================
__global__ __launch_bounds__(256) void gnn_edge_kernel(
    const int* __restrict__ edges, const float* __restrict__ EF,
    const __bf16* __restrict__ H,
    const float* eW0, const float* eB0, const float* eW1, const float* eB1,
    const float* pW0g, const float* pB0g, const float* pW1g, const float* pB1g,
    float* __restrict__ AGG, int E_, int N_) {
  __shared__ alignas(32) __bf16 bEW0[4 * 512];    // K=32(pad)
  __shared__ alignas(32) __bf16 bEW1[8 * 512];    // K=64
  __shared__ alignas(32) __bf16 bPW0[24 * 512];   // K=192
  __shared__ alignas(32) __bf16 bPW1[8 * 512];    // K=64
  __shared__ alignas(32) __bf16 aPk[8][1024];
  __shared__ float sB0e[64], sB1e[64], sB0p[64], sB1p[64];
  int tid = threadIdx.x, nth = blockDim.x;
  int b = blockIdx.y;
  const int*    eb   = edges + (size_t)b * 2 * E_;
  const float*  efb  = EF + (size_t)b * E_ * 16;
  const __bf16* hb   = H + (size_t)b * N_ * 64;
  float*        aggb = AGG + (size_t)b * N_ * 64;

  pack_b(bEW0, eW0, 16, 32, 4, tid, nth);
  pack_b(bEW1, eW1, 64, 64, 4, tid, nth);
  pack_b(bPW0, pW0g, 192, 192, 4, tid, nth);
  pack_b(bPW1, pW1g, 64, 64, 4, tid, nth);
  stage(sB0e, eB0, 64, tid, nth);  stage(sB1e, eB1, 64, tid, nth);
  stage(sB0p, pB0g, 64, tid, nth); stage(sB1p, pB1g, 64, tid, nth);
  __syncthreads();

  int wave = tid >> 5, lane = tid & 31;
  int tile = blockIdx.x * 8 + wave;
  if (tile * 16 >= E_) return;
  int m = lane & 15, hi = lane >> 4, n = m;
  int ebase = tile * 16;
  __bf16* ap = aPk[wave];

  __builtin_prefetch(efb + (size_t)(ebase + 2048) * 16, 0, 0);

  // ---- edge encoder layer 1 ----
  v16bf a0 = load_a_pad16(efb + (size_t)ebase * 16, 16, lane);
#pragma unroll
  for (int nt = 0; nt < 4; ++nt) {
    v8f acc = {};
    acc = wmma_bf16(a0, load_bpk(bEW0, nt, lane), acc);
    float bias = sB0e[nt * 16 + n];
#pragma unroll
    for (int r = 0; r < 8; ++r) {
      float v = acc[r] + bias;
      apack_store(ap, r + 8 * hi, nt * 16 + n, v > 0.0f ? v : 0.0f);
    }
  }
  lds_fence();

  // ---- edge encoder layer 2 (no relu) ----
  v16bf ha0 = load_apk(ap, 0, lane);
  v16bf ha1 = load_apk(ap, 1, lane);
  sched_fence();
#pragma unroll
  for (int nt = 0; nt < 4; ++nt) {
    v8f acc = {};
    acc = wmma_bf16(ha0, load_bpk(bEW1, nt, lane), acc);
    acc = wmma_bf16(ha1, load_bpk(bEW1, 4 + nt, lane), acc);
    float bias = sB1e[nt * 16 + n];
#pragma unroll
    for (int r = 0; r < 8; ++r)
      apack_store(ap, r + 8 * hi, nt * 16 + n, acc[r] + bias);
  }
  lds_fence();

  // ---- gather h rows (bf16, L2-resident): two b128 loads per fragment ----
  int e0m = eb[ebase + m];
  int e1m = eb[E_ + ebase + m];
  const __bf16* h0 = hb + (size_t)e0m * 64;
  const __bf16* h1 = hb + (size_t)e1m * 64;
  v16bf c0 = load_apk(ap, 0, lane);
  v16bf c1 = load_apk(ap, 1, lane);
  v16bf c2 = load_a_bf16row(h0, lane);
  v16bf c3 = load_a_bf16row(h0 + 32, lane);
  v16bf c4 = load_a_bf16row(h1, lane);
  v16bf c5 = load_a_bf16row(h1 + 32, lane);
  sched_fence();

  // ---- edge processor layer 1 (K=192) ----
#pragma unroll
  for (int nt = 0; nt < 4; ++nt) {
    v8f acc = {};
    acc = wmma_bf16(c0, load_bpk(bPW0,      nt, lane), acc);
    acc = wmma_bf16(c1, load_bpk(bPW0,  4 + nt, lane), acc);
    acc = wmma_bf16(c2, load_bpk(bPW0,  8 + nt, lane), acc);
    acc = wmma_bf16(c3, load_bpk(bPW0, 12 + nt, lane), acc);
    acc = wmma_bf16(c4, load_bpk(bPW0, 16 + nt, lane), acc);
    acc = wmma_bf16(c5, load_bpk(bPW0, 20 + nt, lane), acc);
    float bias = sB0p[nt * 16 + n];
#pragma unroll
    for (int r = 0; r < 8; ++r) {
      float v = acc[r] + bias;
      apack_store(ap, r + 8 * hi, nt * 16 + n, v > 0.0f ? v : 0.0f);
    }
  }
  lds_fence();

  // ---- edge processor layer 2 + segment-sum scatter (f32 atomics) ----
  v16bf p0 = load_apk(ap, 0, lane);
  v16bf p1 = load_apk(ap, 1, lane);
  sched_fence();
  int dst[8];
#pragma unroll
  for (int r = 0; r < 8; ++r) dst[r] = eb[ebase + r + 8 * hi];
#pragma unroll
  for (int nt = 0; nt < 4; ++nt) {
    v8f acc = {};
    acc = wmma_bf16(p0, load_bpk(bPW1, nt, lane), acc);
    acc = wmma_bf16(p1, load_bpk(bPW1, 4 + nt, lane), acc);
    float bias = sB1p[nt * 16 + n];
#pragma unroll
    for (int r = 0; r < 8; ++r)
      __hip_atomic_fetch_add(&aggb[(size_t)dst[r] * 64 + nt * 16 + n],
                             acc[r] + bias, __ATOMIC_RELAXED,
                             __HIP_MEMORY_SCOPE_AGENT);
  }
}

// ===========================================================================
// Kernel 3 (fused): node_proc([h|agg]) -> decoder -> residual -> out [N,16]
// ===========================================================================
__global__ __launch_bounds__(256) void gnn_node_kernel(
    const __bf16* __restrict__ H, const float* __restrict__ AGG,
    const float* __restrict__ NF,
    const float* nW0, const float* nB0, const float* nW1, const float* nB1,
    const float* dW0, const float* dB0, const float* dW1, const float* dB1,
    float* __restrict__ OUT, int N_) {
  __shared__ alignas(32) __bf16 bW0[16 * 512];   // K=128, 4 ntiles
  __shared__ alignas(32) __bf16 bW1[8 * 512];    // K=64, 4 ntiles
  __shared__ alignas(32) __bf16 bV0[8 * 512];    // K=64, 4 ntiles
  __shared__ alignas(32) __bf16 bV1[2 * 512];    // K=64, 1 ntile (16 cols)
  __shared__ alignas(32) __bf16 aPk[8][1024];
  __shared__ float b0s[64], b1s[64], c0s[64], c1s[16];
  int tid = threadIdx.x, nth = blockDim.x;
  int b = blockIdx.y;
  const __bf16* hb   = H + (size_t)b * N_ * 64;
  const float*  ab   = AGG + (size_t)b * N_ * 64;
  const float*  nfb  = NF + (size_t)b * N_ * 16;
  float*        outb = OUT + (size_t)b * N_ * 16;

  pack_b(bW0, nW0, 128, 128, 4, tid, nth);
  pack_b(bW1, nW1, 64, 64, 4, tid, nth);
  pack_b(bV0, dW0, 64, 64, 4, tid, nth);
  pack_b(bV1, dW1, 64, 64, 1, tid, nth);
  stage(b0s, nB0, 64, tid, nth); stage(b1s, nB1, 64, tid, nth);
  stage(c0s, dB0, 64, tid, nth); stage(c1s, dB1, 16, tid, nth);
  __syncthreads();

  int wave = tid >> 5, lane = tid & 31;
  int tile = blockIdx.x * 8 + wave;
  if (tile * 16 >= N_) return;
  int m = lane & 15, hi = lane >> 4, n = m;
  __bf16* ap = aPk[wave];
  const __bf16* hrow = hb + (size_t)(tile * 16 + m) * 64;
  const float*  arow = ab + (size_t)(tile * 16 + m) * 64;

  // node_proc layer 1 (K=128: [h | agg])
  v16bf a0 = load_a_bf16row(hrow, lane);
  v16bf a1 = load_a_bf16row(hrow + 32, lane);
  v16bf a2 = load_a_row(arow, lane);
  v16bf a3 = load_a_row(arow + 32, lane);
#pragma unroll
  for (int nt = 0; nt < 4; ++nt) {
    v8f acc = {};
    acc = wmma_bf16(a0, load_bpk(bW0,      nt, lane), acc);
    acc = wmma_bf16(a1, load_bpk(bW0,  4 + nt, lane), acc);
    acc = wmma_bf16(a2, load_bpk(bW0,  8 + nt, lane), acc);
    acc = wmma_bf16(a3, load_bpk(bW0, 12 + nt, lane), acc);
    float bias = b0s[nt * 16 + n];
#pragma unroll
    for (int r = 0; r < 8; ++r) {
      float v = acc[r] + bias;
      apack_store(ap, r + 8 * hi, nt * 16 + n, v > 0.0f ? v : 0.0f);
    }
  }
  lds_fence();

  // node_proc layer 2 (no relu)
  v16bf q0 = load_apk(ap, 0, lane);
  v16bf q1 = load_apk(ap, 1, lane);
  sched_fence();
#pragma unroll
  for (int nt = 0; nt < 4; ++nt) {
    v8f acc = {};
    acc = wmma_bf16(q0, load_bpk(bW1, nt, lane), acc);
    acc = wmma_bf16(q1, load_bpk(bW1, 4 + nt, lane), acc);
    float bias = b1s[nt * 16 + n];
#pragma unroll
    for (int r = 0; r < 8; ++r)
      apack_store(ap, r + 8 * hi, nt * 16 + n, acc[r] + bias);
  }
  lds_fence();

  // decoder layer 1 (relu)
  v16bf r0 = load_apk(ap, 0, lane);
  v16bf r1 = load_apk(ap, 1, lane);
  sched_fence();
#pragma unroll
  for (int nt = 0; nt < 4; ++nt) {
    v8f acc = {};
    acc = wmma_bf16(r0, load_bpk(bV0, nt, lane), acc);
    acc = wmma_bf16(r1, load_bpk(bV0, 4 + nt, lane), acc);
    float bias = c0s[nt * 16 + n];
#pragma unroll
    for (int r = 0; r < 8; ++r) {
      float v = acc[r] + bias;
      apack_store(ap, r + 8 * hi, nt * 16 + n, v > 0.0f ? v : 0.0f);
    }
  }
  lds_fence();

  // decoder layer 2 (64 -> 16) + residual
  v16bf s0 = load_apk(ap, 0, lane);
  v16bf s1 = load_apk(ap, 1, lane);
  sched_fence();
  v8f acc = {};
  acc = wmma_bf16(s0, load_bpk(bV1, 0, lane), acc);
  acc = wmma_bf16(s1, load_bpk(bV1, 1, lane), acc);
  float bias = c1s[n];
#pragma unroll
  for (int r = 0; r < 8; ++r) {
    int row = tile * 16 + r + 8 * hi;
    outb[(size_t)row * 16 + n] = acc[r] + bias + nfb[(size_t)row * 16 + n];
  }
}

// ===========================================================================
extern "C" void kernel_launch(void* const* d_in, const int* in_sizes, int n_in,
                              void* d_out, int out_size, void* d_ws,
                              size_t ws_size, hipStream_t stream) {
  const int B = 2;
  const int E = in_sizes[0] / (2 * B);   // edges: [B,2,E]
  const int N = in_sizes[1] / (16 * B);  // node_features: [B,N,16]

  const int*   edges = (const int*)d_in[0];
  const float* nf    = (const float*)d_in[1];
  const float* ef    = (const float*)d_in[2];
  const float* neW0 = (const float*)d_in[3];
  const float* neB0 = (const float*)d_in[4];
  const float* neW1 = (const float*)d_in[5];
  const float* neB1 = (const float*)d_in[6];
  const float* eeW0 = (const float*)d_in[7];
  const float* eeB0 = (const float*)d_in[8];
  const float* eeW1 = (const float*)d_in[9];
  const float* eeB1 = (const float*)d_in[10];
  const float* epW0 = (const float*)d_in[11];
  const float* epB0 = (const float*)d_in[12];
  const float* epW1 = (const float*)d_in[13];
  const float* epB1 = (const float*)d_in[14];
  const float* npW0 = (const float*)d_in[15];
  const float* npB0 = (const float*)d_in[16];
  const float* npW1 = (const float*)d_in[17];
  const float* npB1 = (const float*)d_in[18];
  const float* dW0  = (const float*)d_in[19];
  const float* dB0  = (const float*)d_in[20];
  const float* dW1  = (const float*)d_in[21];
  const float* dB1  = (const float*)d_in[22];

  size_t hElems = (size_t)B * N * 64;
  __bf16* h  = (__bf16*)d_ws;                                    // bf16 [B,N,64]
  float* agg = (float*)((char*)d_ws +
                        ((hElems * sizeof(__bf16) + 255) & ~(size_t)255));

  long aggN = (long)B * N * 64;
  gnn_zero_kernel<<<(int)((aggN + 255) / 256), 256, 0, stream>>>(agg, aggN);

  int nodeTiles  = (N + 15) / 16;
  int nodeBlocks = (nodeTiles + 7) / 8;
  gnn_enc_kernel<<<dim3(nodeBlocks, B), 256, 0, stream>>>(
      nf, neW0, neB0, neW1, neB1, h, N);

  int edgeTiles  = (E + 15) / 16;
  int edgeBlocks = (edgeTiles + 7) / 8;
  gnn_edge_kernel<<<dim3(edgeBlocks, B), 256, 0, stream>>>(
      edges, ef, h, eeW0, eeB0, eeW1, eeB1, epW0, epB0, epW1, epB1, agg, E, N);

  gnn_node_kernel<<<dim3(nodeBlocks, B), 256, 0, stream>>>(
      h, agg, nf, npW0, npB0, npW1, npB1, dW0, dB0, dW1, dB1, (float*)d_out, N);
}
